// ScaledODENetFE_12987981103717
// MI455X (gfx1250) — matrix operand order repaired and checked by reading, not compiled, not run on hardware
//
#include <hip/hip_runtime.h>

// CDNA5 (gfx1250, wave32) WMMA operand types
typedef __attribute__((ext_vector_type(16))) _Float16 v16h;  // 8 VGPRs: A/B f16 operand
typedef __attribute__((ext_vector_type(8)))  float    v8f;   // 8 VGPRs: C/D f32 operand
typedef __attribute__((ext_vector_type(8)))  unsigned v8u;   // dword view of a v16h

#define SR 44100.0f
constexpr int T     = 44100;
constexpr int BATCH = 32;
constexpr int HID   = 64;
constexpr int ST    = 16;
constexpr int KIN   = 17;   // F + S = 1 + 16 ; padded to K=32, K=17 carries constant 1 -> b1

// pack two f32 -> one dword of two f16 (low = a, high = b): v_cvt_pk_rtz_f16_f32
__device__ __forceinline__ unsigned pk16(float a, float b) {
  return __builtin_bit_cast(unsigned, __builtin_amdgcn_cvt_pkrtz(a, b));
}
// lo<->hi half-wave swap: ds_swizzle_b32, group-of-32, xor_mask=0x10 (SWAPX16)
__device__ __forceinline__ unsigned swap16(unsigned v) {
  return (unsigned)__builtin_amdgcn_ds_swizzle((int)v, 0x401f);
}
// (hiw:low) >> 16 -> picks (low.hi, hiw.lo): v_alignbit_b32
__device__ __forceinline__ unsigned algn(unsigned hiw, unsigned low) {
  return __builtin_amdgcn_alignbit(hiw, low, 16);
}
// CDNA5 has native V_TANH_F32 (transcendental list, ISA flow doc §8.4).
// Fall back to the exp2/rcp formula if this toolchain lacks the builtin.
__device__ __forceinline__ float fast_tanh(float x) {
#if __has_builtin(__builtin_amdgcn_tanhf)
  return __builtin_amdgcn_tanhf(x);               // 1x v_tanh_f32
#else
  float e = __builtin_amdgcn_exp2f(x * 2.8853900817779268f); // 2*log2(e)
  return fmaf(-2.0f, __builtin_amdgcn_rcpf(e + 1.0f), 1.0f);
#endif
}

// One wave (32 threads) integrates 16 batch rows for all 44100 steps.
// Layouts (CDNA5 ISA 7.12.2):
//   A 16x32 f16 : lane = M row;  lo lanes K=0..7 & 16..23, hi lanes K=8..15 & 24..31
//   B 32x16 f16 : lane = N col;  K order as A (this code: A and B both lo lanes K=0..15, hi K=16..31)
//   C/D 16x16 f32: lane = N col; VGPR r = M row r (lo) / r+8 (hi)
__global__ __launch_bounds__(32, 1)
void odenet_fe_kernel(const float* __restrict__ x,  const float* __restrict__ W1,
                      const float* __restrict__ b1, const float* __restrict__ W2,
                      const float* __restrict__ b2, float* __restrict__ out) {
  const int  lane  = threadIdx.x & 31;
  const int  lrow  = lane & 15;
  const bool hi    = lane >= 16;
  const int  bbase = blockIdx.x * 16;           // batch half handled by this wave
  const float inv_sr = 1.0f / SR;

  // ---------- one-time setup: constant weight tiles in A-operand layout ----------
  v16h Wa[4];                                   // W1^T: M=hidden (4 tiles), K=padded input
#pragma unroll
  for (int j = 0; j < 4; ++j) {
    const int m = j * 16 + lrow;
#pragma unroll
    for (int p = 0; p < 16; ++p) {
      const int k = (hi ? 16 : 0) + p;
      float v = (k < KIN) ? W1[k * HID + m] : (k == KIN ? b1[m] : 0.0f);
      Wa[j][p] = (_Float16)v;
    }
  }
  v16h W2a[2];                                  // W2^T: M=state (1 tile), K=hidden (2 tiles)
#pragma unroll
  for (int kt = 0; kt < 2; ++kt) {
#pragma unroll
    for (int p = 0; p < 16; ++p) {
      const int k = kt * 32 + (hi ? 16 : 0) + p;
      W2a[kt][p] = (_Float16)W2[k * ST + lrow];
    }
  }
  v8f cbias;                                    // b2 broadcast in C/D layout
#pragma unroll
  for (int r = 0; r < 8; ++r) cbias[r] = b2[r + (hi ? 8 : 0)];

  v8f yacc;                                     // state y: lane = batch, vgpr = state row
#pragma unroll
  for (int r = 0; r < 8; ++r) yacc[r] = 0.0f;

  if (!hi) out[bbase + lrow] = 0.0f;            // y_full[0] = 0

  // ---------- serial scan: entire recurrence lives in registers ----------
#pragma unroll 1
  for (int t = 0; t < T - 1; ++t) {
    const float xv = x[t * BATCH + bbase + lrow];
    {
      int tp = t + 16; if (tp > T - 1) tp = T - 1;
      __builtin_prefetch(&x[tp * BATCH + bbase + lrow], 0, 3);   // global_prefetch_b8
    }

    // y/sr, packed to f16 pairs (q_i = states (2i, 2i+1) of this lane's half)
    float ysv[8];
#pragma unroll
    for (int r = 0; r < 8; ++r) ysv[r] = yacc[r] * inv_sr;
    unsigned q0 = pk16(ysv[0], ysv[1]);
    unsigned q1 = pk16(ysv[2], ysv[3]);
    unsigned q2 = pk16(ysv[4], ysv[5]);
    unsigned q3 = pk16(ysv[6], ysv[7]);
    // half-wave exchange of the packed state (4 dwords instead of 7 floats)
    unsigned r0 = swap16(q0), r1 = swap16(q1), r2 = swap16(q2), r3 = swap16(q3);

    // inp^T B-tile (K=32 x N=16), lo lanes: K=0..15 = [x, s0..s14] (s8..14 from partner),
    // hi lanes: K=16..31 = [s15, 1(bias->b1), 0...]
    v8u bd;
    bd[0] = hi ? algn(0x3C00u, q3) /* (s15,1.0h) */ : pk16(xv, ysv[0]);
    bd[1] = hi ? 0u : algn(q1, q0);   // (s1,s2)
    bd[2] = hi ? 0u : algn(q2, q1);   // (s3,s4)
    bd[3] = hi ? 0u : algn(q3, q2);   // (s5,s6)
    bd[4] = hi ? 0u : algn(r0, q3);   // (s7,s8)
    bd[5] = hi ? 0u : algn(r1, r0);   // (s9,s10)
    bd[6] = hi ? 0u : algn(r2, r1);   // (s11,s12)
    bd[7] = hi ? 0u : algn(r3, r2);   // (s13,s14)
    v16h bin = __builtin_bit_cast(v16h, bd);

    // GEMM1: h^T tiles (hidden x batch), b1 folded in via the bias-1 input row
    v8f hd[4];
#pragma unroll
    for (int j = 0; j < 4; ++j) {
      v8f zero = {};
      hd[j] = __builtin_amdgcn_wmma_f32_16x16x32_f16(
                  false, Wa[j], false, bin, (short)0, zero, false, false);
    }

    // tanh: 32 elements/lane, ideally 1x v_tanh_f32 each
#pragma unroll
    for (int j = 0; j < 4; ++j)
#pragma unroll
      for (int r = 0; r < 8; ++r) hd[j][r] = fast_tanh(hd[j][r]);

    // h^T D-tiles -> B-tiles for GEMM2: pack to f16 pairs, exchange 4 dwords per K-tile
    v16h bh[2];
#pragma unroll
    for (int kt = 0; kt < 2; ++kt) {
      v8u d;
#pragma unroll
      for (int i = 0; i < 4; ++i) {
        unsigned a  = pk16(hd[2 * kt][2 * i],     hd[2 * kt][2 * i + 1]);     // K=32kt+{2i,2i+1}
        unsigned b  = pk16(hd[2 * kt + 1][2 * i], hd[2 * kt + 1][2 * i + 1]); // K=32kt+16+{...}
        unsigned rc = swap16(hi ? a : b);
        d[i]     = hi ? rc : a;   // halves 0..7  : K = 32kt + (lo: 0..7 | hi: 16..23)
        d[4 + i] = hi ? b : rc;   // halves 8..15 : K = 32kt + (lo: 8..15 | hi: 24..31)
      }
      bh[kt] = __builtin_bit_cast(v16h, d);
    }

    // GEMM2 chained over K with C = y + b2  ->  yacc becomes y_{t+1} directly
    v8f c1;
#pragma unroll
    for (int r = 0; r < 8; ++r) c1[r] = yacc[r] + cbias[r];
    v8f mid = __builtin_amdgcn_wmma_f32_16x16x32_f16(
                  false, W2a[0], false, bh[0], (short)0, c1,  false, false);
    yacc    = __builtin_amdgcn_wmma_f32_16x16x32_f16(
                  false, W2a[1], false, bh[1], (short)0, mid, false, false);

    // output feature 0: state 0 lives in vgpr0 of lo lanes
    if (!hi) out[(t + 1) * BATCH + bbase + lrow] = yacc[0] * inv_sr;
  }
}

extern "C" void kernel_launch(void* const* d_in, const int* in_sizes, int n_in,
                              void* d_out, int out_size, void* d_ws, size_t ws_size,
                              hipStream_t stream) {
  (void)in_sizes; (void)n_in; (void)d_ws; (void)ws_size; (void)out_size;
  const float* x  = (const float*)d_in[0];
  const float* W1 = (const float*)d_in[1];
  const float* b1 = (const float*)d_in[2];
  const float* W2 = (const float*)d_in[3];
  const float* b2 = (const float*)d_in[4];
  float* out = (float*)d_out;
  // 2 independent waves (one per batch half) on separate WGPs; no inter-wave sync.
  hipLaunchKernelGGL(odenet_fe_kernel, dim3(2), dim3(32), 0, stream,
                     x, W1, b1, W2, b2, out);
}
